// ConvLSTMBlock_68779606278332
// MI455X (gfx1250) — matrix operand order repaired and checked by reading, not compiled, not run on hardware
//
#include <hip/hip_runtime.h>
#include <hip/hip_bf16.h>

typedef __attribute__((ext_vector_type(16))) __bf16 v16bf;
typedef __attribute__((ext_vector_type(8)))  __bf16 v8bf;
typedef __attribute__((ext_vector_type(8)))  float  v8f;

#define Bn   4
#define Tn   16
#define Hn   64
#define Wn   64
#define CIN  64
#define CG   256                  // 4 * Cout
#define PSTRIDE 72                // padded cin stride (bank spread), >= 64
#define PSLOTS  (3*66)            // 3 halo rows x 66 halo cols
#define PATCHE  (PSLOTS*64)       // useful elements per path (unpadded count)

// ---------------------------------------------------------------------------
// Pack W and U [3,3,64,256] f32 -> bf16 WMMA B-fragments:
//   layout [tap(9)][ntile(16)][kchunk(2)][lane(32)][elem(16)]
//   B (32x16, 16-bit), this code's k order: lane<16 -> col N=lane&15, K=0..15 ; lane>=16 -> K=16..31
// ---------------------------------------------------------------------------
__global__ __launch_bounds__(256) void pack_weights(const float* __restrict__ W,
                                                    const float* __restrict__ U,
                                                    __bf16* __restrict__ Wp,
                                                    __bf16* __restrict__ Up) {
    int e = blockIdx.x * 256 + threadIdx.x;           // 2 * 147456 threads
    const int per = 9 * 16 * 2 * 512;                 // 147456
    int wsel   = e / per;
    int r      = e - wsel * per;
    int frag   = r >> 9;
    int within = r & 511;
    int lane   = within >> 4;
    int j      = within & 15;
    int tap    = frag >> 5;            // 0..8  (kh*3+kw)
    int ntile  = (frag >> 1) & 15;     // 0..15
    int kc     = frag & 1;             // 0..1
    int n      = ntile * 16 + (lane & 15);
    int cin    = kc * 32 + (lane >> 4) * 16 + j;
    const float* src = wsel ? U : W;
    __bf16*      dst = wsel ? Up : Wp;
    dst[r] = (__bf16)src[(tap * CIN + cin) * CG + n];
}

// ---------------------------------------------------------------------------
// One ConvLSTM timestep: z = conv(x_t, W) + conv(h_prev, U) + b -> gates
// Block = 256 threads (8 waves), owns one full (b, y) image row: M = 64.
// Wave w: ntiles {2w, 2w+1}; m-tiles 0..3 -> 8 independent WMMA acc chains.
// ---------------------------------------------------------------------------
__global__ __launch_bounds__(256) void convlstm_step(
    const float*  __restrict__ x, int t,
    const __bf16* __restrict__ hsrc,
    const __bf16* __restrict__ Wp,
    const __bf16* __restrict__ Up,
    const float*  __restrict__ bias,
    float*        __restrict__ cstate,
    __bf16*       __restrict__ hdst,
    float*        __restrict__ hout)
{
    __shared__ union __align__(16) SM {
        __bf16 patch[2][PSLOTS * PSTRIDE];  // [0]=x, [1]=h  (57024 B)
        float  z[64][CG];                   // 65536 B (reused after WMMA)
    } sm;

    const int tid = threadIdx.x;
    const int bx  = blockIdx.x;            // 256 blocks = B*H rows
    const int b   = bx >> 6;
    const int y   = bx & 63;

    // ---- stage halo patches into LDS (f32->bf16 for x; h already bf16) ----
    for (int idx = tid; idx < PATCHE; idx += 256) {
        int slot = idx >> 6;               // 0..197 : (dyi*66 + col)
        int cin  = idx & 63;
        int dyi  = slot / 66;
        int col  = slot - dyi * 66;
        int gy   = y + dyi - 1;
        int gx   = col - 1;
        float  vx = 0.0f;
        __bf16 vh = (__bf16)0.0f;
        if (gy >= 0 && gy < Hn && gx >= 0 && gx < Wn) {
            vx = x[((((size_t)b * Tn + t) * Hn + gy) * Wn + gx) * CIN + cin];
            vh = hsrc[(((size_t)b * Hn + gy) * Wn + gx) * CIN + cin];
        }
        sm.patch[0][slot * PSTRIDE + cin] = (__bf16)vx;
        sm.patch[1][slot * PSTRIDE + cin] = vh;
    }
    __syncthreads();

    const int lane  = tid & 31;
    const int wv    = tid >> 5;
    const int mrow  = lane & 15;
    const int khalf = lane >> 4;

    v8f acc[4][2] = {};                    // [mtile][ntile] : 8 chains
    for (int path = 0; path < 2; ++path) {
        const __bf16* wp = path ? Up : Wp;
        const __bf16* pp = sm.patch[path];
        #pragma unroll
        for (int tap = 0; tap < 9; ++tap) {
            const int dyi = tap / 3;
            const int dxi = tap - dyi * 3;
            #pragma unroll
            for (int kc = 0; kc < 2; ++kc) {
                // A fragment (16x32 bf16) = two contiguous 16B runs per lane:
                //   elems 0..7  -> K = khalf*16 .. +7       (ds_load_b128)
                //   elems 8..15 -> K = khalf*16 + 8 .. +7   (ds_load_b128)
                v16bf a[4];
                #pragma unroll
                for (int mt = 0; mt < 4; ++mt) {
                    const int m = mt * 16 + mrow;
                    const __bf16* srcp =
                        pp + (dyi * 66 + m + dxi) * PSTRIDE + kc * 32 + khalf * 16;
                    union { v16bf v; v8bf h[2]; } av;
                    av.h[0] = *(const v8bf*)(srcp);
                    av.h[1] = *(const v8bf*)(srcp + 8);
                    a[mt] = av.v;
                }
                #pragma unroll
                for (int nt = 0; nt < 2; ++nt) {
                    const int ntile = wv * 2 + nt;
                    const v16bf bfrag = *(const v16bf*)
                        (wp + ((size_t)((tap * 16 + ntile) * 2 + kc)) * 512 + lane * 16);
                    #pragma unroll
                    for (int mt = 0; mt < 4; ++mt)
                        acc[mt][nt] = __builtin_amdgcn_wmma_f32_16x16x32_bf16(
                            false, a[mt], false, bfrag, (short)0, acc[mt][nt],
                            false, false);
                }
            }
        }
    }
    __syncthreads();   // all patch reads done; LDS union becomes z[][]

    // D layout: vgpr r -> M = r + 8*khalf ; N = lane&15 within n-tile
    #pragma unroll
    for (int mt = 0; mt < 4; ++mt)
        #pragma unroll
        for (int nt = 0; nt < 2; ++nt)
            #pragma unroll
            for (int r = 0; r < 8; ++r) {
                int m = mt * 16 + khalf * 8 + r;
                int n = (wv * 2 + nt) * 16 + mrow;
                sm.z[m][n] = acc[mt][nt][r];
            }
    __syncthreads();

    // ---- fused gating: i,f,c,o (keras order) ----
    for (int e = tid; e < 64 * 64; e += 256) {
        int p = e >> 6;                    // x position within the row
        int c = e & 63;
        float zi = sm.z[p][c]       + bias[c];
        float zf = sm.z[p][64 + c]  + bias[64 + c];
        float zg = sm.z[p][128 + c] + bias[128 + c];
        float zo = sm.z[p][192 + c] + bias[192 + c];
        float ig = fminf(fmaxf(0.2f * zi + 0.5f, 0.0f), 1.0f);
        float fg = fminf(fmaxf(0.2f * zf + 0.5f, 0.0f), 1.0f);
        float gg = tanhf(zg);
        float og = fminf(fmaxf(0.2f * zo + 0.5f, 0.0f), 1.0f);
        size_t gi = (((size_t)b * Hn + y) * Wn + p) * 64 + c;
        float cn = fg * cstate[gi] + ig * gg;
        cstate[gi] = cn;
        float hn = og * tanhf(cn);
        hout[gi] = hn;
        hdst[gi] = (__bf16)hn;
    }
}

// ---------------------------------------------------------------------------
// BatchNorm (inference) + 2x2 maxpool for one timestep slice
// ---------------------------------------------------------------------------
__global__ __launch_bounds__(256) void bn_pool(const float* __restrict__ h,
                                               const float* __restrict__ gamma,
                                               const float* __restrict__ beta,
                                               const float* __restrict__ mean,
                                               const float* __restrict__ var,
                                               float* __restrict__ out, int t)
{
    int idx = blockIdx.x * 256 + threadIdx.x;   // 4*32*32*64 = 262144
    int c  = idx & 63;
    int x2 = (idx >> 6)  & 31;
    int y2 = (idx >> 11) & 31;
    int b  = idx >> 16;
    float sc = gamma[c] * rsqrtf(var[c] + 1e-3f);
    float sh = beta[c] - mean[c] * sc;
    float m = -INFINITY;
    #pragma unroll
    for (int dy = 0; dy < 2; ++dy)
        #pragma unroll
        for (int dx = 0; dx < 2; ++dx) {
            float v = h[(((size_t)b * Hn + (y2 * 2 + dy)) * Wn + (x2 * 2 + dx)) * 64 + c]
                      * sc + sh;
            m = fmaxf(m, v);
        }
    out[((((size_t)b * Tn + t) * 32 + y2) * 32 + x2) * 64 + c] = m;
}

// ---------------------------------------------------------------------------
extern "C" void kernel_launch(void* const* d_in, const int* in_sizes, int n_in,
                              void* d_out, int out_size, void* d_ws, size_t ws_size,
                              hipStream_t stream) {
    (void)in_sizes; (void)n_in; (void)out_size; (void)ws_size;
    const float* x     = (const float*)d_in[0];
    const float* W     = (const float*)d_in[1];
    const float* U     = (const float*)d_in[2];
    const float* bias  = (const float*)d_in[3];
    const float* gamma = (const float*)d_in[4];
    const float* beta  = (const float*)d_in[5];
    const float* mmean = (const float*)d_in[6];
    const float* mvar  = (const float*)d_in[7];
    float* out = (float*)d_out;

    char* ws = (char*)d_ws;
    const size_t WPK = (size_t)147456 * sizeof(__bf16);
    const size_t HB  = (size_t)Bn * Hn * Wn * 64;     // h/c element count
    __bf16* Wp  = (__bf16*)ws;  ws += WPK;
    __bf16* Up  = (__bf16*)ws;  ws += WPK;
    __bf16* hb0 = (__bf16*)ws;  ws += HB * 2;
    __bf16* hb1 = (__bf16*)ws;  ws += HB * 2;
    float*  cst = (float*)ws;   ws += HB * 4;
    float*  hf  = (float*)ws;   ws += HB * 4;

    pack_weights<<<1152, 256, 0, stream>>>(W, U, Wp, Up);
    hipMemsetAsync(hb0, 0, HB * 2, stream);
    hipMemsetAsync(cst, 0, HB * 4, stream);

    for (int t = 0; t < Tn; ++t) {
        const __bf16* hsrc = (t & 1) ? hb1 : hb0;
        __bf16*       hdst = (t & 1) ? hb0 : hb1;
        convlstm_step<<<256, 256, 0, stream>>>(x, t, hsrc, Wp, Up, bias, cst, hdst, hf);
        bn_pool<<<1024, 256, 0, stream>>>(hf, gamma, beta, mmean, mvar, out, t);
    }
}
